// GiantGraphGCN_74483322847875
// MI455X (gfx1250) — compile-verified, matrix operand order checked by reading, not run on hardware
//
#include <hip/hip_runtime.h>

// ---------------------------------------------------------------------------
// GiantGraphGCN for MI455X (gfx1250, wave32, WMMA).
// Dense matmuls: v_wmma_f32_16x16x32_bf16 (bf16 in / f32 accumulate),
// f32->bf16 conversion fused into the LDS staging, fragments read from LDS
// as 16-byte ds_load_b128 chunks (layout-exact, conflict-free 80B stride),
// global->register prefetch pipelined across K-steps (no branchy guards:
// out-of-range rows are clamped and masked only at the epilogue store).
// Graph message passing: f32 global atomics (memory-bound part of workload).
// ---------------------------------------------------------------------------

typedef __bf16 bhalf;
typedef __attribute__((ext_vector_type(16))) __bf16 v16bf;
typedef __attribute__((ext_vector_type(8)))  __bf16 v8bf;
typedef __attribute__((ext_vector_type(4)))  __bf16 v4bf;
typedef __attribute__((ext_vector_type(8)))  float  v8f;

static constexpr int kND    = 4000;
static constexpr int kNPROT = 19000;
static constexpr int kFD    = 1024;
static constexpr int kP0    = 128;     // EP + FP
static constexpr int kD     = 256;
static constexpr int kC1    = 384;     // D + 2K
static constexpr int kEDP   = 80000;
static constexpr int kEPP   = 400000;

static constexpr int BM = 128, BN = 64, BK = 32;
// LDS row stride in bf16 units: 80 bytes. Keeps rows 16B-aligned for b128
// reads and makes the 16 lanes' 16B fragment reads hit 16 disjoint 4-bank
// groups (20 banks/row shift, gcd(20,64)=4 -> all 16 rows distinct).
static constexpr int LSTR = 40;

// ---------------------------------------------------------------------------
// GEMM: C[M,N] = op( A[M,K] @ W[K,N] + bias ), bf16 WMMA, fp32 accumulate.
// flags bit0: relu. Row-major f32 A/W/C with leading dims lda/ldw/ldc.
// REQUIRES: K % 32 == 0, N % 64 == 0, lda/ldw % 4 == 0, M >= 1 (true here:
// K in {128,256,384,1024}, N in {256,384}).
// 256 threads = 8 waves arranged 4(M) x 2(N); each wave owns a 32x32 tile
// (2x2 16x16 WMMA accumulators). Block tile 128x64, K-step 32.
// Software pipeline: global float4 loads for tile k+1 are issued right after
// the barrier publishing tile k, and consumed (cvt+ds_store) next iteration,
// so their s_wait_loadcnt lands after this iteration's WMMAs.
// ---------------------------------------------------------------------------
__global__ __launch_bounds__(256) void gg_gemm_wmma(
    const float* __restrict__ A, int lda,
    const float* __restrict__ W, int ldw,
    const float* __restrict__ bias,
    float* __restrict__ C, int ldc,
    int M, int N, int K, int flags)
{
  __shared__ __align__(16) bhalf As[BM][LSTR];    // [m][k]
  __shared__ __align__(16) bhalf Bst[BN][LSTR];   // transposed: [n][k]

  const int tid  = threadIdx.x;
  const int wave = tid >> 5;
  const int lane = tid & 31;
  const int wm   = wave & 3;        // wave row: 32 rows each
  const int wn   = wave >> 2;       // wave col: 32 cols each
  const int g    = lane >> 4;       // lane group (0/1)
  const int lm   = lane & 15;       // lane % 16

  const int block_m = blockIdx.x * BM;
  const int block_n = blockIdx.y * BN;

  // Per-thread staging coordinates (constant across the K loop).
  const int qa = tid & 7;           // quad-of-4-floats along K for A
  const int ra = tid >> 3;          // A row 0..31, +32 per unrolled it
  const int qb = tid & 15;          // quad-of-4-floats along N for B
  const int rb = tid >> 4;          // B k-row 0..15, +16 per unrolled it

  // Hoisted row base pointers; rows clamped to M-1 (no branches — results
  // for clamped rows are finite duplicates masked at the epilogue store).
  const float* arow[4];
  #pragma unroll
  for (int it = 0; it < 4; ++it) {
    int gm = block_m + ra + it * 32;
    if (gm >= M) gm = M - 1;
    arow[it] = A + (size_t)gm * lda + 4 * qa;
  }
  const float* brow[2];
  #pragma unroll
  for (int it = 0; it < 2; ++it)
    brow[it] = W + (size_t)(rb + it * 16) * ldw + block_n + 4 * qb;

  v8f acc00 = {0.f,0.f,0.f,0.f,0.f,0.f,0.f,0.f};
  v8f acc01 = acc00, acc10 = acc00, acc11 = acc00;

  // Prefetch tile 0 into registers.
  float4 aPre[4], bPre[2];
  #pragma unroll
  for (int it = 0; it < 4; ++it) aPre[it] = *(const float4*)(arow[it]);
  #pragma unroll
  for (int it = 0; it < 2; ++it) bPre[it] = *(const float4*)(brow[it]);

  for (int k0 = 0; k0 < K; k0 += BK) {
    // ---- publish current tile to LDS (f32 -> bf16)
    #pragma unroll
    for (int it = 0; it < 4; ++it) {
      float4 v = aPre[it];
      v4bf h;
      h[0] = (bhalf)v.x; h[1] = (bhalf)v.y; h[2] = (bhalf)v.z; h[3] = (bhalf)v.w;
      *(v4bf*)(&As[ra + it * 32][4 * qa]) = h;  // 8B aligned ds_store
    }
    #pragma unroll
    for (int it = 0; it < 2; ++it) {
      float4 v = bPre[it];
      int r = rb + it * 16;
      Bst[4 * qb + 0][r] = (bhalf)v.x;
      Bst[4 * qb + 1][r] = (bhalf)v.y;
      Bst[4 * qb + 2][r] = (bhalf)v.z;
      Bst[4 * qb + 3][r] = (bhalf)v.w;
    }
    __syncthreads();

    // ---- issue next tile's global loads (consumed next iteration)
    int k1 = k0 + BK;
    if (k1 < K) {
      #pragma unroll
      for (int it = 0; it < 4; ++it)
        aPre[it] = *(const float4*)(arow[it] + k1);
      size_t boff = (size_t)k1 * ldw;
      #pragma unroll
      for (int it = 0; it < 2; ++it)
        bPre[it] = *(const float4*)(brow[it] + boff);
    }

    // ---- fragments: per ISA 16-bit 16x32 layout, a lane's v16bf is two
    // contiguous 8-element (16B) runs: k = 8g+0..7 and k = 16+8g+0..7.
    const bhalf* a0 = &As [wm * 32 + lm     ][0];
    const bhalf* a1 = &As [wm * 32 + 16 + lm][0];
    const bhalf* b0 = &Bst[wn * 32 + lm     ][0];
    const bhalf* b1 = &Bst[wn * 32 + 16 + lm][0];
    v8bf a0l = *(const v8bf*)(a0 + 8 * g), a0h = *(const v8bf*)(a0 + 16 + 8 * g);
    v8bf a1l = *(const v8bf*)(a1 + 8 * g), a1h = *(const v8bf*)(a1 + 16 + 8 * g);
    v8bf b0l = *(const v8bf*)(b0 + 8 * g), b0h = *(const v8bf*)(b0 + 16 + 8 * g);
    v8bf b1l = *(const v8bf*)(b1 + 8 * g), b1h = *(const v8bf*)(b1 + 16 + 8 * g);
    v16bf af0 = __builtin_shufflevector(a0l, a0h, 0,1,2,3,4,5,6,7,8,9,10,11,12,13,14,15);
    v16bf af1 = __builtin_shufflevector(a1l, a1h, 0,1,2,3,4,5,6,7,8,9,10,11,12,13,14,15);
    v16bf bf0 = __builtin_shufflevector(b0l, b0h, 0,1,2,3,4,5,6,7,8,9,10,11,12,13,14,15);
    v16bf bf1 = __builtin_shufflevector(b1l, b1h, 0,1,2,3,4,5,6,7,8,9,10,11,12,13,14,15);

    acc00 = __builtin_amdgcn_wmma_f32_16x16x32_bf16(false, af0, false, bf0,
                                                    (short)0, acc00, false, false);
    acc01 = __builtin_amdgcn_wmma_f32_16x16x32_bf16(false, af0, false, bf1,
                                                    (short)0, acc01, false, false);
    acc10 = __builtin_amdgcn_wmma_f32_16x16x32_bf16(false, af1, false, bf0,
                                                    (short)0, acc10, false, false);
    acc11 = __builtin_amdgcn_wmma_f32_16x16x32_bf16(false, af1, false, bf1,
                                                    (short)0, acc11, false, false);
    __syncthreads();
  }

  // ---- epilogue: C/D layout: n = lane%16, m = v + 8*(lane/16)
  const bool relu = (flags & 1) != 0;
  #pragma unroll
  for (int j = 0; j < 2; ++j) {
    int gn = block_n + wn * 32 + j * 16 + lm;   // always < N (N % 64 == 0)
    float bv = bias ? bias[gn] : 0.f;
    #pragma unroll
    for (int i = 0; i < 2; ++i) {
      v8f acc = (i == 0) ? (j == 0 ? acc00 : acc01)
                         : (j == 0 ? acc10 : acc11);
      #pragma unroll
      for (int v = 0; v < 8; ++v) {
        int gm = block_m + wm * 32 + i * 16 + v + 8 * g;
        if (gm >= M) continue;
        float r = acc[v] + bv;
        if (relu) r = fmaxf(r, 0.f);
        C[(size_t)gm * ldc + gn] = r;
      }
    }
  }
}

// ---------------------------------------------------------------------------
// h_prot = concat(prot_emb[19000x64], x_prots[19000x64]) -> [19000x128]
// ---------------------------------------------------------------------------
__global__ void gg_build_hprot(const float* __restrict__ emb,
                               const float* __restrict__ xp,
                               float* __restrict__ hp)
{
  int i = blockIdx.x * blockDim.x + threadIdx.x;
  if (i >= kNPROT * kP0) return;
  int r = i >> 7, c = i & 127;
  hp[i] = (c < 64) ? emb[r * 64 + c] : xp[r * 64 + (c - 64)];
}

// ---------------------------------------------------------------------------
// Edge scatter-add: dst[didx[e], :width] += src[sidx[e], :width]
// One 64-thread (2 wave32) block per edge.
// ---------------------------------------------------------------------------
__global__ __launch_bounds__(64) void gg_scatter_add(
    const float* __restrict__ src, int lds_,
    const int* __restrict__ sidx, const int* __restrict__ didx,
    float* __restrict__ dst, int ldd_, int width)
{
  int e = blockIdx.x;
  int s = sidx[e], d = didx[e];
  const float* sp = src + (size_t)s * lds_;
  float*       dp = dst + (size_t)d * ldd_;
  for (int j = threadIdx.x; j < width; j += 64)
    atomicAdd(&dp[j], sp[j]);
}

// ---------------------------------------------------------------------------
// Attention reduction: M64[64,64] += V^T Z ; cuv[0:64] += colsum(U),
// cuv[64:128] += colsum(V).  t layout: [N,256] = [U|V|Z|T].
// Each thread owns 16 (k,j) register accumulators; atomics at the end.
// ---------------------------------------------------------------------------
__global__ __launch_bounds__(256) void gg_att_reduce(
    const float* __restrict__ t, int Nrows,
    float* __restrict__ M64, float* __restrict__ cuv)
{
  __shared__ float sU[64], sV[64], sZ[64];
  float accM[16];
  #pragma unroll
  for (int p = 0; p < 16; ++p) accM[p] = 0.f;
  float accU = 0.f, accV = 0.f;

  const int k  = threadIdx.x >> 2;
  const int j0 = threadIdx.x & 3;

  int rpb = (Nrows + gridDim.x - 1) / gridDim.x;
  int r0 = blockIdx.x * rpb;
  int r1 = r0 + rpb; if (r1 > Nrows) r1 = Nrows;

  for (int r = r0; r < r1; ++r) {
    const float* tr = t + (size_t)r * 256;
    if (threadIdx.x < 64)       sU[threadIdx.x]       = tr[threadIdx.x];
    else if (threadIdx.x < 128) sV[threadIdx.x - 64]  = tr[threadIdx.x];
    else if (threadIdx.x < 192) sZ[threadIdx.x - 128] = tr[threadIdx.x];
    __syncthreads();
    float vk = sV[k];
    #pragma unroll
    for (int p = 0; p < 16; ++p) accM[p] += vk * sZ[j0 + 4 * p];
    if (threadIdx.x < 64) { accU += sU[threadIdx.x]; accV += sV[threadIdx.x]; }
    __syncthreads();
  }
  #pragma unroll
  for (int p = 0; p < 16; ++p) atomicAdd(&M64[k * 64 + j0 + 4 * p], accM[p]);
  if (threadIdx.x < 64) {
    atomicAdd(&cuv[threadIdx.x], accU);
    atomicAdd(&cuv[64 + threadIdx.x], accV);
  }
}

// norm = dot(colsum(U), colsum(V)) / N + 1e-6   (single 64-thread block)
__global__ void gg_att_norm(const float* __restrict__ cuv,
                            float* __restrict__ norm, int Nrows)
{
  __shared__ float red[64];
  int i = threadIdx.x;
  red[i] = cuv[i] * cuv[64 + i];
  __syncthreads();
  for (int s = 32; s > 0; s >>= 1) {
    if (i < s) red[i] += red[i + s];
    __syncthreads();
  }
  if (i == 0) *norm = red[0] / (float)Nrows + 1e-6f;
}

// out[r, off+0:64] = (U[r,:] @ M64) / norm ; out[r, off+64:128] = T[r,:]
// 8 rows per 256-thread block; M64 staged in LDS.
__global__ __launch_bounds__(256) void gg_att_apply(
    const float* __restrict__ t, int Nrows,
    const float* __restrict__ M64, const float* __restrict__ normp,
    float* __restrict__ out, int ldo, int coloff)
{
  __shared__ float sM[64 * 64];
  __shared__ float sU[8][64];
  for (int i = threadIdx.x; i < 64 * 64; i += 256) sM[i] = M64[i];
  int r0 = blockIdx.x * 8;
  for (int i = threadIdx.x; i < 8 * 64; i += 256) {
    int rl = i >> 6, c = i & 63;
    int r = r0 + rl;
    sU[rl][c] = (r < Nrows) ? t[(size_t)r * 256 + c] : 0.f;
  }
  __syncthreads();
  float inv = 1.0f / (*normp);
  #pragma unroll
  for (int w = 0; w < 4; ++w) {
    int idx = threadIdx.x + w * 256;
    int rl = idx >> 7;
    int c  = idx & 127;
    int r = r0 + rl;
    if (r >= Nrows) continue;
    if (c < 64) {
      float s = 0.f;
      #pragma unroll
      for (int kk = 0; kk < 64; ++kk) s += sU[rl][kk] * sM[kk * 64 + c];
      out[(size_t)r * ldo + coloff + c] = s * inv;
    } else {
      out[(size_t)r * ldo + coloff + c] = t[(size_t)r * 256 + 192 + (c - 64)];
    }
  }
}

__global__ void gg_relu(float* __restrict__ x, int n)
{
  int i = blockIdx.x * blockDim.x + threadIdx.x;
  if (i < n) x[i] = fmaxf(x[i], 0.f);
}

// out[r, 0:384] = relu(hd1[r,:] + od2[r,:]) with ldo=512
__global__ void gg_residual_relu(const float* __restrict__ a,
                                 const float* __restrict__ b,
                                 float* __restrict__ out, int nrows)
{
  int i = blockIdx.x * blockDim.x + threadIdx.x;
  if (i >= nrows * kC1) return;
  int r = i / kC1, c = i % kC1;
  out[(size_t)r * 512 + c] = fmaxf(a[i] + b[i], 0.f);
}

// ---------------------------------------------------------------------------
extern "C" void kernel_launch(void* const* d_in, const int* in_sizes, int n_in,
                              void* d_out, int out_size, void* d_ws, size_t ws_size,
                              hipStream_t stream)
{
  (void)in_sizes; (void)n_in; (void)out_size; (void)ws_size;
  // Inputs in setup_inputs() dict order (params flattened recursively):
  const float* x_drugs  = (const float*)d_in[0];   // [4000,1024]
  const float* x_prots  = (const float*)d_in[1];   // [19000,64]
  const float* prot_emb = (const float*)d_in[2];   // [19000,64]
  const float* c1_wdself = (const float*)d_in[3];  // [1024,256]
  const float* c1_wpself = (const float*)d_in[4];  // [128,256]
  const float* c1_wd2p   = (const float*)d_in[5];
  const float* c1_wp2d   = (const float*)d_in[6];
  const float* c1_wp2p   = (const float*)d_in[7];
  const float* c1_bd     = (const float*)d_in[8];
  const float* c1_bp     = (const float*)d_in[9];
  const float* ra_wdself = (const float*)d_in[10]; // [384,384]
  const float* ra_wpself = (const float*)d_in[11];
  const float* ra_wd2p   = (const float*)d_in[12];
  const float* ra_wp2d   = (const float*)d_in[13];
  const float* ra_wp2p   = (const float*)d_in[14];
  const float* ra_bd     = (const float*)d_in[15];
  const float* ra_bp     = (const float*)d_in[16];
  const float* rb_wdself = (const float*)d_in[17];
  const float* rb_wp2d   = (const float*)d_in[20];
  const float* rb_bd     = (const float*)d_in[22];
  const float* ad0_w = (const float*)d_in[24];     // [1024,256]
  const float* ad0_b = (const float*)d_in[25];
  const float* ap0_w = (const float*)d_in[26];     // [128,256]
  const float* ap0_b = (const float*)d_in[27];
  const float* ad1_w = (const float*)d_in[28];     // [384,256]
  const float* ad1_b = (const float*)d_in[29];
  const int* dpi_src = (const int*)d_in[32];
  const int* dpi_dst = (const int*)d_in[33];
  const int* ppi_src = (const int*)d_in[34];
  const int* ppi_dst = (const int*)d_in[35];
  float* out = (float*)d_out;                      // [4000,512]

  // Workspace layout (~165 MB of f32)
  float* p = (float*)d_ws;
  float* h_prot = p; p += (size_t)kNPROT * kP0;
  float* t_att  = p; p += (size_t)kNPROT * 256;
  float* msgP1  = p; p += (size_t)kNPROT * kC1;
  float* msgP2  = p; p += (size_t)kNPROT * kC1;
  float* msgD   = p; p += (size_t)kND * kC1;
  float* hd1    = p; p += (size_t)kND * kC1;
  float* hp1    = p; p += (size_t)kNPROT * kC1;
  float* odr    = p; p += (size_t)kND * kC1;
  float* opr    = p; p += (size_t)kNPROT * kC1;
  float* od2    = p; p += (size_t)kND * kC1;
  float* attM   = p; p += 64 * 64;
  float* cuv    = p; p += 128;
  float* normp  = p; p += 1;

  auto gemm = [&](const float* A, int lda, const float* W, int ldw,
                  const float* bias, float* C, int ldc,
                  int M, int N, int K, int flags) {
    dim3 grid((M + BM - 1) / BM, (N + BN - 1) / BN);
    gg_gemm_wmma<<<grid, 256, 0, stream>>>(A, lda, W, ldw, bias, C, ldc, M, N, K, flags);
  };
  auto attention = [&](const float* X, int lda, int K, const float* wA,
                       const float* bA, int Nrows, float* dstc, int ldo, int coloff) {
    gemm(X, lda, wA, 256, bA, t_att, 256, Nrows, 256, K, 1 /*relu*/);
    hipMemsetAsync(attM, 0, (64 * 64 + 128) * sizeof(float), stream);
    gg_att_reduce<<<256, 256, 0, stream>>>(t_att, Nrows, attM, cuv);
    gg_att_norm<<<1, 64, 0, stream>>>(cuv, normp, Nrows);
    gg_att_apply<<<(Nrows + 7) / 8, 256, 0, stream>>>(t_att, Nrows, attM, normp,
                                                      dstc, ldo, coloff);
  };

  // h_prot = concat(prot_emb, x_prots)
  gg_build_hprot<<<(kNPROT * kP0 + 255) / 256, 256, 0, stream>>>(prot_emb, x_prots, h_prot);

  // -------- conv1 + attention 0 --------
  gemm(x_drugs, kFD, c1_wd2p, kD, nullptr, msgD, kD, kND, kD, kFD, 0);      // drug->prot msgs
  gemm(h_prot, kP0, c1_wp2d, kD, nullptr, msgP1, kD, kNPROT, kD, kP0, 0);   // prot->drug msgs
  gemm(h_prot, kP0, c1_wp2p, kD, nullptr, msgP2, kD, kNPROT, kD, kP0, 0);   // prot->prot msgs
  gemm(x_drugs, kFD, c1_wdself, kD, c1_bd, hd1, kC1, kND, kD, kFD, 0);      // hd1[:, :256]
  gemm(h_prot, kP0, c1_wpself, kD, c1_bp, hp1, kC1, kNPROT, kD, kP0, 0);    // hp1[:, :256]
  gg_scatter_add<<<kEDP, 64, 0, stream>>>(msgP1, kD, dpi_dst, dpi_src, hd1, kC1, kD); // p2d
  gg_scatter_add<<<kEDP, 64, 0, stream>>>(msgD,  kD, dpi_src, dpi_dst, hp1, kC1, kD); // d2p
  gg_scatter_add<<<kEPP, 64, 0, stream>>>(msgP2, kD, ppi_src, ppi_dst, hp1, kC1, kD); // p2p
  attention(x_drugs, kFD, kFD, ad0_w, ad0_b, kND, hd1, kC1, kD);            // hd1[:, 256:]
  attention(h_prot, kP0, kP0, ap0_w, ap0_b, kNPROT, hp1, kC1, kD);          // hp1[:, 256:]
  gg_relu<<<(kND * kC1 + 255) / 256, 256, 0, stream>>>(hd1, kND * kC1);
  gg_relu<<<(kNPROT * kC1 + 255) / 256, 256, 0, stream>>>(hp1, kNPROT * kC1);

  // -------- res_a --------
  gemm(hd1, kC1, ra_wd2p, kC1, nullptr, msgD, kC1, kND, kC1, kC1, 0);
  gemm(hp1, kC1, ra_wp2d, kC1, nullptr, msgP1, kC1, kNPROT, kC1, kC1, 0);
  gemm(hp1, kC1, ra_wp2p, kC1, nullptr, msgP2, kC1, kNPROT, kC1, kC1, 0);
  gemm(hd1, kC1, ra_wdself, kC1, ra_bd, odr, kC1, kND, kC1, kC1, 0);
  gemm(hp1, kC1, ra_wpself, kC1, ra_bp, opr, kC1, kNPROT, kC1, kC1, 0);
  gg_scatter_add<<<kEDP, 64, 0, stream>>>(msgP1, kC1, dpi_dst, dpi_src, odr, kC1, kC1);
  gg_scatter_add<<<kEDP, 64, 0, stream>>>(msgD,  kC1, dpi_src, dpi_dst, opr, kC1, kC1);
  gg_scatter_add<<<kEPP, 64, 0, stream>>>(msgP2, kC1, ppi_src, ppi_dst, opr, kC1, kC1);
  gg_relu<<<(kND * kC1 + 255) / 256, 256, 0, stream>>>(odr, kND * kC1);
  gg_relu<<<(kNPROT * kC1 + 255) / 256, 256, 0, stream>>>(opr, kNPROT * kC1);

  // -------- res_b (protein output of res_b is dead code in the reference) --------
  gemm(odr, kC1, rb_wdself, kC1, rb_bd, od2, kC1, kND, kC1, kC1, 0);
  gemm(opr, kC1, rb_wp2d, kC1, nullptr, msgP1, kC1, kNPROT, kC1, kC1, 0);
  gg_scatter_add<<<kEDP, 64, 0, stream>>>(msgP1, kC1, dpi_dst, dpi_src, od2, kC1, kC1);

  // -------- output: relu(hd1 + od2) || attention(hd1, att_d1) --------
  gg_residual_relu<<<(kND * kC1 + 255) / 256, 256, 0, stream>>>(hd1, od2, out, kND);
  attention(hd1, kC1, kC1, ad1_w, ad1_b, kND, out, 512, kC1);
}
